// DeepFM_61761629717133
// MI455X (gfx1250) — compile-verified, matrix operand order checked by reading, not run on hardware
//
#include <hip/hip_runtime.h>
#include <hip/hip_bf16.h>
#include <math.h>

typedef unsigned short u16;
typedef __attribute__((ext_vector_type(16))) __bf16 v16bf;
typedef __attribute__((ext_vector_type(8)))  float  v8f;

#define BATCH   16384
#define EMB     64
#define ONEHOT  26
#define MULTIHOT 2
#define MH_LEN  20
#define DENSE   13
#define K0PAD   1824   // 28*64 + 13 -> padded to 57*32
#define N0      1024
#define N1      512

__device__ __forceinline__ u16 f2bf(float x) {
    unsigned u = __float_as_uint(x);
    unsigned r = (u + 0x7FFFu + ((u >> 16) & 1u)) >> 16;
    return (u16)r;
}
__device__ __forceinline__ float bf2f(u16 x) {
    return __uint_as_float(((unsigned)x) << 16);
}

// ---------------- weight repack: (K x N) f32 -> (N x Kpad) bf16, K zero-padded --
__global__ void repack_w_kernel(const float* __restrict__ w, u16* __restrict__ wt,
                                int K, int Kpad, int N) {
    int id = blockIdx.x * blockDim.x + threadIdx.x;
    if (id >= N * Kpad) return;
    int n = id / Kpad, k = id - n * Kpad;
    wt[id] = (k < K) ? f2bf(w[(size_t)k * N + n]) : (u16)0;
}

// ---------------- embed + FM kernel: one block (256 thr) per batch row ---------
__global__ void embed_fm_kernel(const float* __restrict__ dense,
                                const int*   __restrict__ onehot,
                                const int*   __restrict__ multihot,
                                const float* __restrict__ fm_w,
                                const float* __restrict__ fm_emb,
                                u16*  __restrict__ A0,     // (BATCH x K0PAD) bf16
                                float* __restrict__ fm_out) {
    __shared__ float ssum[EMB];
    __shared__ float ssq[EMB];
    __shared__ float sscal[2];     // [0]=fm1st  [1]=fm2nd
    const int row = blockIdx.x;
    const int t = threadIdx.x;
    const int e = t & 63;          // embedding dim
    const int g = t >> 6;          // lane-group 0..3
    if (t < EMB) { ssum[t] = 0.f; ssq[t] = 0.f; }
    if (t < 2) sscal[t] = 0.f;
    __syncthreads();

    u16* arow = A0 + (size_t)row * K0PAD;
    float ls = 0.f, lq = 0.f;

    // onehot fields striped over groups
    for (int f = g; f < ONEHOT; f += 4) {
        int idx = onehot[row * ONEHOT + f];
        float v = fm_emb[(size_t)idx * EMB + e];
        ls += v; lq += v * v;
        arow[f * EMB + e] = f2bf(v);
    }
    // multihot mean combiner: groups 2,3 own slots 0,1
    if (g >= 2) {
        int m = g - 2;
        float acc = 0.f;
        const int* mh = multihot + ((size_t)row * MULTIHOT + m) * MH_LEN;
        #pragma unroll 4
        for (int j = 0; j < MH_LEN; ++j)
            acc += fm_emb[(size_t)mh[j] * EMB + e];
        float v = acc * (1.0f / MH_LEN);
        ls += v; lq += v * v;
        arow[(ONEHOT + m) * EMB + e] = f2bf(v);
    }
    atomicAdd(&ssum[e], ls);
    atomicAdd(&ssq[e], lq);

    // FM first order (onehot only, as in reference)
    if (t < ONEHOT)
        atomicAdd(&sscal[0], fm_w[onehot[row * ONEHOT + t]]);

    // dense tail + zero pad: cols 1792..1823
    if (t < 32) {
        float v = (t < DENSE) ? dense[row * DENSE + t] : 0.0f;
        arow[28 * EMB + t] = f2bf(v);
    }
    __syncthreads();
    if (t < EMB) {
        float s = ssum[t];
        atomicAdd(&sscal[1], 0.5f * (s * s - ssq[t]));
    }
    __syncthreads();
    if (t == 0) fm_out[row] = sscal[0] + sscal[1];
}

// ---------------- bf16 WMMA GEMM + bias + ReLU --------------------------------
// C(M x N) = relu(A(M x K) * Bt^T + bias), A row-major bf16, Bt is (N x K) bf16.
// Block: 256 threads = 8 waves. Block tile: 256(M) x 64(N); wave tile 32 x 64.
// Per 32-K step, per wave: 2 A-frags + 4 B-frags (12 ds_load_b128) -> 8 WMMAs.
#define LROW 40   // padded LDS row stride in u16 (80B) to spread banks

union FragCast { uint4 u[2]; v16bf v; };

__device__ __forceinline__ v16bf load_frag(const u16* base, int half) {
    // lane<16: K{0..7, 16..23}; lane>=16: K{8..15, 24..31}
    FragCast f;
    f.u[0] = *(const uint4*)(base + half * 8);
    f.u[1] = *(const uint4*)(base + 16 + half * 8);
    return f.v;
}

template <int N, int K>
__global__ __launch_bounds__(256, 1)
void gemm_bf16_relu_kernel(const u16* __restrict__ A,
                           const u16* __restrict__ Bt,
                           const float* __restrict__ bias,
                           u16* __restrict__ C) {
    __shared__ u16 lA[256 * LROW];   // 20 KB
    __shared__ u16 lB[64 * LROW];    //  5 KB
    const int t = threadIdx.x;
    const int wave = t >> 5, lane = t & 31;
    const int half = lane >> 4, l15 = lane & 15;
    const int M0 = blockIdx.y * 256;
    const int Nb = blockIdx.x * 64;

    v8f acc[2][4];
    #pragma unroll
    for (int i = 0; i < 2; ++i)
        #pragma unroll
        for (int j = 0; j < 4; ++j)
            acc[i][j] = (v8f){0.f,0.f,0.f,0.f,0.f,0.f,0.f,0.f};

    // hoisted per-thread global pointers (strength-reduced: += 32 elems per step)
    const u16* pA[4];
    u16* sA[4];
    #pragma unroll
    for (int i = 0; i < 4; ++i) {
        int c = t + 256 * i;            // A tile: 1024 x 16B chunks
        int r = c >> 2, s = c & 3;
        pA[i] = A + (size_t)(M0 + r) * K + s * 8;
        sA[i] = lA + r * LROW + s * 8;
    }
    const u16* pB;
    u16* sB;
    {
        int r = t >> 2, s = t & 3;      // B tile: 256 x 16B chunks
        pB = Bt + (size_t)(Nb + r) * K + s * 8;
        sB = lB + r * LROW + s * 8;
    }
    const u16* fragA0 = lA + (wave * 32 + l15) * LROW;
    const u16* fragA1 = fragA0 + 16 * LROW;
    const u16* fragB  = lB + l15 * LROW;

    #pragma unroll 1
    for (int k0 = 0; k0 < K; k0 += 32) {
        // stage A tile (coalesced 16B chunks) + prefetch next tile (streamed)
        #pragma unroll
        for (int i = 0; i < 4; ++i) {
            uint4 d = *(const uint4*)pA[i];
            __builtin_prefetch(pA[i] + 32, 0, 3);
            *(uint4*)sA[i] = d;
            pA[i] += 32;
        }
        // stage B tile (L2-resident repacked weights)
        {
            uint4 d = *(const uint4*)pB;
            *(uint4*)sB = d;
            pB += 32;
        }
        __syncthreads();

        v16bf a0 = load_frag(fragA0, half);
        v16bf a1 = load_frag(fragA1, half);
        #pragma unroll
        for (int j = 0; j < 4; ++j) {
            v16bf b = load_frag(fragB + j * 16 * LROW, half);
            acc[0][j] = __builtin_amdgcn_wmma_f32_16x16x32_bf16(
                false, a0, false, b, (short)0, acc[0][j], false, false);
            acc[1][j] = __builtin_amdgcn_wmma_f32_16x16x32_bf16(
                false, a1, false, b, (short)0, acc[1][j], false, false);
        }
        __syncthreads();
    }

    // Epilogue: D layout -> lanes 0-15: M=r, N=lane; lanes 16-31: M=8+r, N=lane-16
    const float bv = bias[0];
    #pragma unroll
    for (int i = 0; i < 2; ++i) {
        const int mrow = M0 + wave * 32 + i * 16 + half * 8;
        #pragma unroll
        for (int j = 0; j < 4; ++j) {
            const int col = Nb + j * 16 + l15;
            #pragma unroll
            for (int r = 0; r < 8; ++r) {
                float v = acc[i][j][r] + bv;
                v = v > 0.f ? v : 0.f;
                C[(size_t)(mrow + r) * N + col] = f2bf(v);
            }
        }
    }
}

// ---------------- final: GEMV(512->1) + relu + FM + sigmoid -------------------
__global__ void final_kernel(const u16* __restrict__ C1,
                             const float* __restrict__ w2,
                             const float* __restrict__ b2,
                             const float* __restrict__ fm,
                             float* __restrict__ out) {
    const int t = threadIdx.x;
    const int wave = t >> 5, lane = t & 31;
    const int row = blockIdx.x * 8 + wave;
    const u16* crow = C1 + (size_t)row * N1;
    float acc = 0.f;
    #pragma unroll
    for (int k = lane; k < N1; k += 32)
        acc += bf2f(crow[k]) * w2[k];
    #pragma unroll
    for (int off = 16; off > 0; off >>= 1)
        acc += __shfl_xor(acc, off, 32);
    if (lane == 0) {
        float nn = acc + b2[0];
        nn = nn > 0.f ? nn : 0.f;
        float x = fm[row] + nn;
        out[row] = 1.0f / (1.0f + __expf(-x));
    }
}

// ---------------- launch ------------------------------------------------------
static inline size_t align256(size_t x) { return (x + 255) & ~(size_t)255; }

extern "C" void kernel_launch(void* const* d_in, const int* in_sizes, int n_in,
                              void* d_out, int out_size, void* d_ws, size_t ws_size,
                              hipStream_t stream) {
    const float* dense    = (const float*)d_in[0];
    const int*   onehot   = (const int*)  d_in[1];
    const int*   multihot = (const int*)  d_in[2];
    const float* fm_w     = (const float*)d_in[3];
    const float* fm_emb   = (const float*)d_in[4];
    const float* w0       = (const float*)d_in[5];
    const float* b0       = (const float*)d_in[6];
    const float* w1       = (const float*)d_in[7];
    const float* b1       = (const float*)d_in[8];
    const float* w2       = (const float*)d_in[9];
    const float* b2       = (const float*)d_in[10];
    float* out = (float*)d_out;

    char* ws = (char*)d_ws;
    size_t off = 0;
    u16* A0  = (u16*)(ws + off); off = align256(off + (size_t)BATCH * K0PAD * 2);
    u16* C0  = (u16*)(ws + off); off = align256(off + (size_t)BATCH * N0 * 2);
    u16* C1  = (u16*)(ws + off); off = align256(off + (size_t)BATCH * N1 * 2);
    u16* Wt0 = (u16*)(ws + off); off = align256(off + (size_t)N0 * K0PAD * 2);
    u16* Wt1 = (u16*)(ws + off); off = align256(off + (size_t)N1 * N0 * 2);
    float* fmbuf = (float*)(ws + off); off = align256(off + (size_t)BATCH * 4);

    // 1) repack weights to bf16, transposed (N x Kpad)
    {
        int tot0 = N0 * K0PAD;
        repack_w_kernel<<<(tot0 + 255) / 256, 256, 0, stream>>>(w0, Wt0, 1805, K0PAD, N0);
        int tot1 = N1 * N0;
        repack_w_kernel<<<(tot1 + 255) / 256, 256, 0, stream>>>(w1, Wt1, N0, N0, N1);
    }
    // 2) embedding gathers + FM + fused activation build (bf16)
    embed_fm_kernel<<<BATCH, 256, 0, stream>>>(dense, onehot, multihot, fm_w, fm_emb,
                                               A0, fmbuf);
    // 3) layer 0: (16384 x 1824) x (1824 x 1024) -> relu -> bf16
    gemm_bf16_relu_kernel<N0, K0PAD>
        <<<dim3(N0 / 64, BATCH / 256), 256, 0, stream>>>(A0, Wt0, b0, C0);
    // 4) layer 1: (16384 x 1024) x (1024 x 512) -> relu -> bf16
    gemm_bf16_relu_kernel<N1, N0>
        <<<dim3(N1 / 64, BATCH / 256), 256, 0, stream>>>(C0, Wt1, b1, C1);
    // 5) layer 2 (GEMV) + FM combine + sigmoid
    final_kernel<<<BATCH / 8, 256, 0, stream>>>(C1, w2, b2, fmbuf, out);
}
